// HybridGraphModel_40888088658019
// MI455X (gfx1250) — matrix operand order repaired and checked by reading, compiled, not run on hardware
//
#include <hip/hip_runtime.h>

// ---------------------------------------------------------------------------
// CDNA5 (gfx1250, wave32) implementation of the hybrid GNN:
//   SplineConv -> SAGE -> SplineConv -> SAGE -> concat -> Linear
// Dense GEMMs via v_wmma_f32_16x16x32_bf16 (bf16 in, f32 accumulate), with the
// shared B matrix staged into LDS via GLOBAL_LOAD_ASYNC_TO_LDS_B128 (ASYNCcnt).
// Edge aggregation via wave-per-edge gather + global f32 atomics.
// ---------------------------------------------------------------------------

typedef __attribute__((ext_vector_type(16))) __bf16 v16bf;
typedef __attribute__((ext_vector_type(8)))  float  v8f;

#define C64 64
#define KMATS 25
#define XW_STRIDE (KMATS * C64)   // 1600 elements per node row in XW
#define BS_STRIDE 72              // padded LDS row stride (elems): 144B = 9*16B

// ---- bf16 helpers (bit-exact RNE, no reliance on native __bf16 arithmetic) --
__device__ __forceinline__ __bf16 f2bf(float f) {
  union { float f; unsigned u; } a; a.f = f;
  unsigned r = a.u + 0x7FFFu + ((a.u >> 16) & 1u);
  unsigned short h = (unsigned short)(r >> 16);
  return __builtin_bit_cast(__bf16, h);
}
__device__ __forceinline__ float bfbits2f(unsigned short h) {
  union { unsigned u; float f; } a; a.u = ((unsigned)h) << 16;
  return a.f;
}

// ---- WMMA fragment loaders --------------------------------------------------
// A-matrix 16x32 bf16 (contiguous k order, same as B): lane half h=lane>>4, row M=lane&15.
// Per-lane 16 elems: K = h*16 + {0..15}  (+ chunk*32).
__device__ __forceinline__ v16bf load_a_frag(const __bf16* base, int row, int koff) {
  union { v16bf v; uint4 q[2]; } f;
  const uint4* p = reinterpret_cast<const uint4*>(base + (size_t)row * C64 + koff);
  f.q[0] = p[0];   // K = koff .. koff+7
  f.q[1] = p[1];   // K = koff+8 .. koff+15
  return f.v;
}
// B-matrix 32x16 bf16 (contiguous k order, same as A): lane half h,
// column N = lane&15, per-lane 16 elems: K = h*16 + 0..15 (+ chunk*32).
// Read from the padded LDS tile (row-transposed B, contiguous in K).
__device__ __forceinline__ v16bf load_b_frag_lds(const __bf16* bs, int nrow, int koff16) {
  union { v16bf v; uint4 q[2]; } f;
  const uint4* p = reinterpret_cast<const uint4*>(bs + nrow * BS_STRIDE + koff16);
  f.q[0] = p[0];
  f.q[1] = p[1];
  return f.v;
}

// ---------------------------------------------------------------------------
// Generic 64-wide WMMA GEMM:  out = act( A[M,64] @ B[64,64] (+bias)
//                                        (+addend*rowScale) )
// One wave computes a 16x64 tile (4 WMMA accumulators, 2 K-chunks of 32).
// gridDim.y = batch (B and out-batch offset advance per batch).
// B (8KB, shared by all waves/tiles) is staged to LDS with async loads.
// ---------------------------------------------------------------------------
__global__ void k_gemm64(const __bf16* __restrict__ A,
                         const __bf16* __restrict__ Bt,     // [batch][64][64] transposed
                         int tiles,
                         float* __restrict__ outF,          // optional f32 out
                         __bf16* __restrict__ outB,         // optional bf16 out
                         long long outRowStride,            // in elements
                         long long outBatchOff,             // in elements
                         const float* __restrict__ bias,    // [64] or null
                         const float* __restrict__ addend,  // [M,64] f32 or null
                         const float* __restrict__ degScale,// [M] -> addend /= max(deg,1)
                         int relu)
{
  __shared__ __bf16 Bs[BS_STRIDE * C64];     // padded 64x64 bf16 tile (9216 B)

  const int batch = blockIdx.y;
  const __bf16* B = Bt + (size_t)batch * (C64 * C64);

  // ---- async stage of B into LDS: 64 rows x 128B -> padded rows of 144B ----
  // GLOBAL_LOAD_ASYNC_TO_LDS_B128: per-lane 16B memory->LDS, tracked by ASYNCcnt.
  // The LDS address is obtained by ptrtoint of &Bs[...]: the low 32 bits of a
  // generic pointer to LDS are the LDS byte address (ISA aperture rule), and
  // the ptrtoint captures Bs so the asm "memory" clobber legally writes it
  // (otherwise LLVM proves Bs is never stored and folds its reads to undef).
  for (int i = threadIdx.x; i < 512; i += blockDim.x) {   // 512 x 16B = 8KB
    int row = i >> 3, seg = i & 7;
    unsigned lds_addr = (unsigned)(uintptr_t)(Bs + row * BS_STRIDE + seg * 8);
    const void* gaddr = (const char*)B + row * 128 + seg * 16;
    asm volatile("global_load_async_to_lds_b128 %0, %1, off"
                 :: "v"(lds_addr), "v"(gaddr) : "memory");
  }
  asm volatile("s_wait_asynccnt 0x0" ::: "memory");   // this wave's copies done
  __syncthreads();                                    // all waves' copies visible

  const int wavesPerBlock = blockDim.x >> 5;
  const int tileM = blockIdx.x * wavesPerBlock + (threadIdx.x >> 5);
  if (tileM < tiles) {                       // wave-uniform: EXEC all-ones inside
    const int lane = threadIdx.x & 31;
    const int lh = lane >> 4;                // lane half
    const int ll = lane & 15;

    v8f acc[4];
#pragma unroll
    for (int i = 0; i < 4; ++i) acc[i] = (v8f){0.f,0.f,0.f,0.f,0.f,0.f,0.f,0.f};

    const int rowA = tileM * 16 + ll;
#pragma unroll
    for (int chunk = 0; chunk < 2; ++chunk) {
      v16bf a = load_a_frag(A, rowA, lh * 16 + chunk * 32);
#pragma unroll
      for (int nt = 0; nt < 4; ++nt) {
        v16bf b = load_b_frag_lds(Bs, nt * 16 + ll, lh * 16 + chunk * 32);
        acc[nt] = __builtin_amdgcn_wmma_f32_16x16x32_bf16(
            false, a, false, b, (short)0, acc[nt], false, false);
      }
    }

    // Epilogue. D layout: VGPR v holds M = tileM*16 + lh*8 + v, N = nt*16 + ll.
#pragma unroll
    for (int nt = 0; nt < 4; ++nt) {
      const int col = nt * 16 + ll;
      const float bc = bias ? bias[col] : 0.0f;
#pragma unroll
      for (int v = 0; v < 8; ++v) {
        const int row = tileM * 16 + lh * 8 + v;
        float val = acc[nt][v] + bc;
        if (addend) {
          float scl = 1.0f;
          if (degScale) scl = 1.0f / fmaxf(degScale[row], 1.0f);
          val += addend[(size_t)row * C64 + col] * scl;
        }
        if (relu) val = fmaxf(val, 0.0f);
        const size_t oi = (size_t)row * outRowStride + (size_t)batch * outBatchOff + col;
        if (outF) outF[oi] = val;
        if (outB) outB[oi] = f2bf(val);
      }
    }
  }
}

// ---- small utility kernels --------------------------------------------------
__global__ void k_zero(float* p, int n) {
  int i = blockIdx.x * blockDim.x + threadIdx.x;
  if (i < n) p[i] = 0.0f;
}

__global__ void k_cast_bf16(const float* __restrict__ in, __bf16* __restrict__ out, int n) {
  int i = blockIdx.x * blockDim.x + threadIdx.x;
  if (i < n) out[i] = f2bf(in[i]);
}

// transpose-convert weights: Wt[b][o][i] = (bf16) W[b][i][o]   (64x64 blocks)
__global__ void k_transpose_bf16(const float* __restrict__ W, __bf16* __restrict__ Wt) {
  const float* w = W + (size_t)blockIdx.x * (C64 * C64);
  __bf16* wt = Wt + (size_t)blockIdx.x * (C64 * C64);
  for (int idx = threadIdx.x; idx < C64 * C64; idx += blockDim.x) {
    int o = idx >> 6, i = idx & 63;
    wt[idx] = f2bf(w[i * C64 + o]);
  }
}

// degree-1 open B-spline basis, kernel_size=5, dim=2 -> 4 (kidx, weight) per edge
__global__ void k_basis(const float* __restrict__ attr, float* __restrict__ bw,
                        int* __restrict__ kid, int E) {
  int e = blockIdx.x * blockDim.x + threadIdx.x;
  if (e >= E) return;
  float v0 = attr[2 * e]     * 4.0f;
  float v1 = attr[2 * e + 1] * 4.0f;
  float lo0 = fminf(fmaxf(floorf(v0), 0.0f), 4.0f);
  float lo1 = fminf(fmaxf(floorf(v1), 0.0f), 4.0f);
  float f0 = v0 - lo0, f1 = v1 - lo1;
  int i0 = (int)lo0, i1 = (int)lo1;
  int h0 = min(i0 + 1, 4), h1 = min(i1 + 1, 4);
  int t = 0;
#pragma unroll
  for (int s0 = 0; s0 < 2; ++s0) {
#pragma unroll
    for (int s1 = 0; s1 < 2; ++s1) {
      float b0 = s0 ? f0 : 1.0f - f0;
      float b1 = s1 ? f1 : 1.0f - f1;
      int k = (s0 ? h0 : i0) + 5 * (s1 ? h1 : i1);
      bw[e * 4 + t] = b0 * b1;
      kid[e * 4 + t] = k;
      ++t;
    }
  }
}

__global__ void k_degree(const int* __restrict__ dst, float* __restrict__ deg, int E) {
  int e = blockIdx.x * blockDim.x + threadIdx.x;
  if (e < E) atomicAdd(&deg[dst[e]], 1.0f);
}

// spline edge pass: one wave per edge, 2 channels per lane.
// gathers 4 bf16 rows of XW[src, kidx] (each row = one 128B line), scatters f32.
__global__ void k_spline_edges(const __bf16* __restrict__ XW,
                               const int* __restrict__ src, const int* __restrict__ dst,
                               const float* __restrict__ bw, const int* __restrict__ kid,
                               float* __restrict__ agg, int E)
{
  int wave = blockIdx.x * (blockDim.x >> 5) + (threadIdx.x >> 5);
  int lane = threadIdx.x & 31;
  if (wave >= E) return;
  const int s = src[wave], d = dst[wave];
  const float4 w4 = *reinterpret_cast<const float4*>(bw + (size_t)wave * 4);
  const int4   k4 = *reinterpret_cast<const int4*>(kid + (size_t)wave * 4);
  const float wgt[4] = {w4.x, w4.y, w4.z, w4.w};
  const int   kk [4] = {k4.x, k4.y, k4.z, k4.w};
  const size_t base = (size_t)s * XW_STRIDE + lane * 2;
  float a0 = 0.0f, a1 = 0.0f;
#pragma unroll
  for (int t = 0; t < 4; ++t) {
    unsigned u = *reinterpret_cast<const unsigned*>(XW + base + kk[t] * C64);
    a0 += wgt[t] * bfbits2f((unsigned short)(u & 0xFFFFu));
    a1 += wgt[t] * bfbits2f((unsigned short)(u >> 16));
  }
  atomicAdd(&agg[(size_t)d * C64 + lane * 2],     a0);
  atomicAdd(&agg[(size_t)d * C64 + lane * 2 + 1], a1);
}

// SAGE edge pass: one wave per edge, gather x[src] f32 and scatter-add.
__global__ void k_sage_edges(const float* __restrict__ x,
                             const int* __restrict__ src, const int* __restrict__ dst,
                             float* __restrict__ agg, int E)
{
  int wave = blockIdx.x * (blockDim.x >> 5) + (threadIdx.x >> 5);
  int lane = threadIdx.x & 31;
  if (wave >= E) return;
  const int s = src[wave], d = dst[wave];
  const float2 v = *reinterpret_cast<const float2*>(x + (size_t)s * C64 + lane * 2);
  atomicAdd(&agg[(size_t)d * C64 + lane * 2],     v.x);
  atomicAdd(&agg[(size_t)d * C64 + lane * 2 + 1], v.y);
}

// mean + cast:  out_bf16 = agg / max(deg,1)
__global__ void k_mean_cast(const float* __restrict__ agg, const float* __restrict__ deg,
                            __bf16* __restrict__ out, int n)
{
  int i = blockIdx.x * blockDim.x + threadIdx.x;
  if (i >= n) return;
  float dv = fmaxf(deg[i >> 6], 1.0f);
  out[i] = f2bf(agg[i] / dv);
}

// ---------------------------------------------------------------------------
extern "C" void kernel_launch(void* const* d_in, const int* in_sizes, int n_in,
                              void* d_out, int out_size, void* d_ws, size_t ws_size,
                              hipStream_t stream)
{
  const float* x       = (const float*)d_in[0];
  const int*   eidx    = (const int*)d_in[1];
  const float* eattr   = (const float*)d_in[2];
  const float* w0      = (const float*)d_in[3];
  const float* root0   = (const float*)d_in[4];
  const float* b0      = (const float*)d_in[5];
  const float* l1_w    = (const float*)d_in[6];
  const float* l1_b    = (const float*)d_in[7];
  const float* r1_w    = (const float*)d_in[8];
  const float* w2      = (const float*)d_in[9];
  const float* root2   = (const float*)d_in[10];
  const float* b2      = (const float*)d_in[11];
  const float* l3_w    = (const float*)d_in[12];
  const float* l3_b    = (const float*)d_in[13];
  const float* r3_w    = (const float*)d_in[14];
  const float* final_w = (const float*)d_in[15];
  const float* final_b = (const float*)d_in[16];

  const int N = in_sizes[0] / C64;      // 50000
  const int E = in_sizes[1] / 2;        // 800000
  const int* src = eidx;
  const int* dst = eidx + E;

  // ---- workspace carve-out (256B aligned) ----
  char* wsb = (char*)d_ws; size_t off = 0;
  auto alloc = [&](size_t bytes) -> void* {
    void* p = wsb + off; off += (bytes + 255) & ~(size_t)255; return p;
  };
  __bf16* XW    = (__bf16*)alloc((size_t)N * XW_STRIDE * 2);   // 160 MB
  float*  agg   = (float*) alloc((size_t)N * C64 * 4);
  float*  deg   = (float*) alloc((size_t)N * 4);
  float*  x1    = (float*) alloc((size_t)N * C64 * 4);
  float*  x2    = (float*) alloc((size_t)N * C64 * 4);
  float*  x3    = (float*) alloc((size_t)N * C64 * 4);
  float*  tmp   = (float*) alloc((size_t)N * C64 * 4);
  __bf16* xb0   = (__bf16*)alloc((size_t)N * C64 * 2);
  __bf16* xb1   = (__bf16*)alloc((size_t)N * C64 * 2);
  __bf16* xb2   = (__bf16*)alloc((size_t)N * C64 * 2);
  __bf16* xb3   = (__bf16*)alloc((size_t)N * C64 * 2);
  __bf16* xb4   = (__bf16*)alloc((size_t)N * C64 * 2);
  __bf16* meanb = (__bf16*)alloc((size_t)N * C64 * 2);
  float*  bw    = (float*) alloc((size_t)E * 4 * 4);
  int*    kid   = (int*)   alloc((size_t)E * 4 * 4);
  __bf16* w0t   = (__bf16*)alloc((size_t)KMATS * C64 * C64 * 2);
  __bf16* w2t   = (__bf16*)alloc((size_t)KMATS * C64 * C64 * 2);
  __bf16* r0t   = (__bf16*)alloc((size_t)C64 * C64 * 2);
  __bf16* r2t   = (__bf16*)alloc((size_t)C64 * C64 * 2);
  __bf16* l1t   = (__bf16*)alloc((size_t)C64 * C64 * 2);
  __bf16* rw1t  = (__bf16*)alloc((size_t)C64 * C64 * 2);
  __bf16* l3t   = (__bf16*)alloc((size_t)C64 * C64 * 2);
  __bf16* rw3t  = (__bf16*)alloc((size_t)C64 * C64 * 2);
  __bf16* fwt   = (__bf16*)alloc((size_t)5 * C64 * C64 * 2);
  (void)ws_size; (void)n_in; (void)out_size;

  float* out = (float*)d_out;

  const int NC = N * C64;
  const int tiles = (N + 15) / 16;                 // 3125 (exact for N=50000)
  dim3 gemmBlk(128);                               // 4 waves / block
  auto gemmGrid = [&](int batch) { return dim3((tiles + 3) / 4, batch); };

  // ---- weight prep: transpose + bf16 ----
  k_transpose_bf16<<<dim3(KMATS), 256, 0, stream>>>(w0, w0t);
  k_transpose_bf16<<<dim3(KMATS), 256, 0, stream>>>(w2, w2t);
  k_transpose_bf16<<<dim3(1), 256, 0, stream>>>(root0, r0t);
  k_transpose_bf16<<<dim3(1), 256, 0, stream>>>(root2, r2t);
  k_transpose_bf16<<<dim3(1), 256, 0, stream>>>(l1_w, l1t);
  k_transpose_bf16<<<dim3(1), 256, 0, stream>>>(r1_w, rw1t);
  k_transpose_bf16<<<dim3(1), 256, 0, stream>>>(l3_w, l3t);
  k_transpose_bf16<<<dim3(1), 256, 0, stream>>>(r3_w, rw3t);
  k_transpose_bf16<<<dim3(5), 256, 0, stream>>>(final_w, fwt);   // [320,64] = 5 blocks

  // ---- graph-invariant precompute ----
  k_cast_bf16<<<(NC + 255) / 256, 256, 0, stream>>>(x, xb0, NC);
  k_basis<<<(E + 255) / 256, 256, 0, stream>>>(eattr, bw, kid, E);
  k_zero<<<(N + 255) / 256, 256, 0, stream>>>(deg, N);
  k_degree<<<(E + 255) / 256, 256, 0, stream>>>(dst, deg, E);

  const dim3 edgeGrid((E + 7) / 8), edgeBlk(256);  // 8 waves/block, 1 wave/edge

  // ================= layer 0: SplineConv(x -> x1) =================
  k_gemm64<<<gemmGrid(KMATS), gemmBlk, 0, stream>>>(
      xb0, w0t, tiles, nullptr, XW, XW_STRIDE, C64,
      nullptr, nullptr, nullptr, 0);
  k_zero<<<(NC + 255) / 256, 256, 0, stream>>>(agg, NC);
  k_spline_edges<<<edgeGrid, edgeBlk, 0, stream>>>(XW, src, dst, bw, kid, agg, E);
  k_gemm64<<<gemmGrid(1), gemmBlk, 0, stream>>>(
      xb0, r0t, tiles, x1, xb1, C64, 0, b0, agg, deg, 1);

  // ================= layer 1: SAGEConv(x1 -> x2) =================
  k_zero<<<(NC + 255) / 256, 256, 0, stream>>>(agg, NC);
  k_sage_edges<<<edgeGrid, edgeBlk, 0, stream>>>(x1, src, dst, agg, E);
  k_mean_cast<<<(NC + 255) / 256, 256, 0, stream>>>(agg, deg, meanb, NC);
  k_gemm64<<<gemmGrid(1), gemmBlk, 0, stream>>>(
      xb1, rw1t, tiles, tmp, nullptr, C64, 0, nullptr, nullptr, nullptr, 0);
  k_gemm64<<<gemmGrid(1), gemmBlk, 0, stream>>>(
      meanb, l1t, tiles, x2, xb2, C64, 0, l1_b, tmp, nullptr, 1);

  // ================= layer 2: SplineConv(x2 -> x3) =================
  k_gemm64<<<gemmGrid(KMATS), gemmBlk, 0, stream>>>(
      xb2, w2t, tiles, nullptr, XW, XW_STRIDE, C64,
      nullptr, nullptr, nullptr, 0);
  k_zero<<<(NC + 255) / 256, 256, 0, stream>>>(agg, NC);
  k_spline_edges<<<edgeGrid, edgeBlk, 0, stream>>>(XW, src, dst, bw, kid, agg, E);
  k_gemm64<<<gemmGrid(1), gemmBlk, 0, stream>>>(
      xb2, r2t, tiles, x3, xb3, C64, 0, b2, agg, deg, 1);

  // ================= layer 3: SAGEConv(x3 -> x4) =================
  k_zero<<<(NC + 255) / 256, 256, 0, stream>>>(agg, NC);
  k_sage_edges<<<edgeGrid, edgeBlk, 0, stream>>>(x3, src, dst, agg, E);
  k_mean_cast<<<(NC + 255) / 256, 256, 0, stream>>>(agg, deg, meanb, NC);
  k_gemm64<<<gemmGrid(1), gemmBlk, 0, stream>>>(
      xb3, rw3t, tiles, tmp, nullptr, C64, 0, nullptr, nullptr, nullptr, 0);
  k_gemm64<<<gemmGrid(1), gemmBlk, 0, stream>>>(
      meanb, l3t, tiles, nullptr, xb4, C64, 0, l3_b, tmp, nullptr, 1);

  // ========= final Linear(concat(x,x1..x4) @ W + b) as 5 accumulated GEMMs =========
  k_gemm64<<<gemmGrid(1), gemmBlk, 0, stream>>>(
      xb0, fwt + 0 * C64 * C64, tiles, out, nullptr, C64, 0, final_b, nullptr, nullptr, 0);
  k_gemm64<<<gemmGrid(1), gemmBlk, 0, stream>>>(
      xb1, fwt + 1 * C64 * C64, tiles, out, nullptr, C64, 0, nullptr, out, nullptr, 0);
  k_gemm64<<<gemmGrid(1), gemmBlk, 0, stream>>>(
      xb2, fwt + 2 * C64 * C64, tiles, out, nullptr, C64, 0, nullptr, out, nullptr, 0);
  k_gemm64<<<gemmGrid(1), gemmBlk, 0, stream>>>(
      xb3, fwt + 3 * C64 * C64, tiles, out, nullptr, C64, 0, nullptr, out, nullptr, 0);
  k_gemm64<<<gemmGrid(1), gemmBlk, 0, stream>>>(
      xb4, fwt + 4 * C64 * C64, tiles, out, nullptr, C64, 0, nullptr, out, nullptr, 0);
}
